// NuGraph3_model_3719441679017
// MI455X (gfx1250) — compile-verified, matrix operand order checked by reading, not run on hardware
//
#include <hip/hip_runtime.h>
#include <hip/hip_bf16.h>

// ---------------- problem constants (match reference) ----------------
#define NHIT   30000
#define NEDGE  90000
#define NSP    30000
#define HD     128
#define NCLASS 5
#define NITERS 3
#define WAVES  8   // 256-thread blocks = 8 wave32s

typedef __attribute__((ext_vector_type(16))) __bf16        v16bf;
typedef __attribute__((ext_vector_type(8)))  float         v8f;
typedef __attribute__((ext_vector_type(8)))  unsigned int  v8u;

static __device__ __forceinline__ unsigned short f32_to_bf16(float f) {
    unsigned int u = __float_as_uint(f);
    u += 0x7FFFu + ((u >> 16) & 1u);   // round-to-nearest-even
    return (unsigned short)(u >> 16);
}

// k-pair slot inside a 32-wide K chunk for the 16-bit A/B fragment layout
// (ISA 7.12.2: dword i<4 covers K = half*8 + 2i, dword i>=4 covers K = 16 + half*8 + 2(i-4))
static __device__ __forceinline__ int kmap(int i, int half) {
    return (i < 4) ? (half * 8 + 2 * i) : (16 + half * 8 + 2 * (i - 4));
}

// Inverse of kmap: dword index within a fragment for even k. Fragment-order LDS layout:
// dword index = kc*256 + (half*16 + row)*8 + i   (per 32-wide K chunk: 32 lanes x 8 dwords)
static __device__ __forceinline__ int frag_idx(int row, int k /*even*/) {
    int kc = k >> 5, kk = k & 31;
    int half = (kk >= 16) ? ((kk - 16) >> 3) : (kk >> 3);
    int i    = (kk >= 16) ? (4 + (((kk - 16) & 7) >> 1)) : ((kk & 7) >> 1);
    return kc * 256 + (half * 16 + row) * 8 + i;
}

// One wave computes a [16 x 128] f32 tile from fragment-order operands:
//   A: bf16 in LDS, fragment-major (2x ds_load_b128 per chunk)
//   W: bf16 in global, fragment-major (2x global_load_b128 per chunk x n-tile)
template <int KTOT>
static __device__ __forceinline__ void wmma_frag_tile(const unsigned int* __restrict__ sA,
                                                      const v8u* __restrict__ Wf,
                                                      int lane, v8f* acc) {
    for (int kc = 0; kc < KTOT / 32; ++kc) {
        v8u au = *(const v8u*)(sA + (kc * 32 + lane) * 8);
        v16bf av = __builtin_bit_cast(v16bf, au);
#pragma unroll
        for (int nt = 0; nt < 8; ++nt) {
            v8u bu = Wf[(kc * 8 + nt) * 32 + lane];
            v16bf bv = __builtin_bit_cast(v16bf, bu);
            acc[nt] = __builtin_amdgcn_wmma_f32_16x16x32_bf16(
                false, av, false, bv, (short)0, acc[nt], false, false);
        }
    }
}

// ---------------- weight packing: f32 [K x 128] -> bf16 fragment-major dwords ----------------
// Output dword index = ((kc*8 + nt)*32 + lane)*8 + i; total K*64 dwords.
__global__ void pack_w_kernel(const float* __restrict__ W, unsigned int* __restrict__ Wf, int K) {
    int idx = blockIdx.x * blockDim.x + threadIdx.x;
    int total = K * 64;
    if (idx >= total) return;
    int i = idx & 7, lane = (idx >> 3) & 31, nt = (idx >> 8) & 7, kc = idx >> 11;
    int half = lane >> 4, col = nt * 16 + (lane & 15);
    int k = kc * 32 + kmap(i, half);
    unsigned int lo = f32_to_bf16(W[k * HD + col]);
    unsigned int hi = f32_to_bf16(W[(k + 1) * HD + col]);
    Wf[idx] = lo | (hi << 16);
}

// ---------------- encoder: h = relu(x @ W_enc + b_enc), K=4 ----------------
__global__ void encoder_kernel(const float* __restrict__ x, const float* __restrict__ W,
                               const float* __restrict__ b, float* __restrict__ h, int N) {
    int i = blockIdx.x * blockDim.x + threadIdx.x;
    if (i >= N * HD) return;
    int n = i / HD, c = i % HD;
    float a = b[c];
#pragma unroll
    for (int k = 0; k < 4; ++k) a += x[n * 4 + k] * W[k * HD + c];
    h[i] = a > 0.f ? a : 0.f;
}

// ---------------- segment counts ----------------
__global__ void count_kernel(const int* __restrict__ idx, float* __restrict__ cnt, int M) {
    int i = blockIdx.x * blockDim.x + threadIdx.x;
    if (i < M) atomicAdd(&cnt[idx[i]], 1.0f);
}

// ---------------- edge MLP + scatter into agg ----------------
__global__ __launch_bounds__(256, 1) void edge_mlp_kernel(
    const float* __restrict__ h, const int* __restrict__ edge,
    const v8u* __restrict__ W1f, const float* __restrict__ b1,
    const v8u* __restrict__ W2f, const float* __restrict__ b2,
    float* __restrict__ agg, int E) {
    __shared__ unsigned int sA[WAVES][256 * 8];   // K=256 fragment-order bf16 pairs (8KB/wave)
    __shared__ unsigned int sM[WAVES][128 * 8];   // K=128 fragment-order hidden acts (4KB/wave)
    const int wave = threadIdx.x >> 5, lane = threadIdx.x & 31;
    const int tile = blockIdx.x * WAVES + wave;
    const long e0  = (long)tile * 16;
    const bool active = e0 < E;
    __builtin_prefetch(W1f, 0, 1);
    int dsti = 0, srci = 0;
    if (active) {
        if (lane < 16) {
            long e = e0 + lane;
            if (e >= E) e = E - 1;
            srci = edge[e];            // row 0 = src
            dsti = edge[(long)E + e];  // row 1 = dst
        }
        // stage concat(h[dst], h[src]) as bf16 in fragment order; 1024 float4 chunks
        for (int t = lane; t < 16 * 64; t += 32) {
            int row = t >> 6, c4 = (t & 63) << 2;
            int d = __shfl(dsti, row);
            int s = __shfl(srci, row);
            const float4 v = (c4 < HD) ? *(const float4*)&h[(long)d * HD + c4]
                                       : *(const float4*)&h[(long)s * HD + (c4 - HD)];
            unsigned int p0 = (unsigned int)f32_to_bf16(v.x) | ((unsigned int)f32_to_bf16(v.y) << 16);
            unsigned int p1 = (unsigned int)f32_to_bf16(v.z) | ((unsigned int)f32_to_bf16(v.w) << 16);
            int fi = frag_idx(row, c4);
            *(uint2*)&sA[wave][fi] = make_uint2(p0, p1);
        }
    }
    __syncthreads();
    if (!active) return;

    const int col = lane & 15, half = lane >> 4;
    v8f acc[8] = {};
    wmma_frag_tile<256>(sA[wave], W1f, lane, acc);           // layer 1
    unsigned short* sMs = (unsigned short*)sM[wave];
#pragma unroll
    for (int nt = 0; nt < 8; ++nt) {
#pragma unroll
        for (int i = 0; i < 8; ++i) {
            int m = i + 8 * half, n = nt * 16 + col;
            float v = acc[nt][i] + b1[n];
            v = v > 0.f ? v : 0.f;
            int fi = frag_idx(m, n & ~1);
            sMs[fi * 2 + (n & 1)] = f32_to_bf16(v);
        }
    }
    v8f acc2[8] = {};
    wmma_frag_tile<128>(sM[wave], W2f, lane, acc2);          // layer 2
#pragma unroll
    for (int i = 0; i < 8; ++i) {
        int m  = i + 8 * half;
        int d  = __shfl(dsti, m);
        bool ok = (e0 + m) < E;
#pragma unroll
        for (int nt = 0; nt < 8; ++nt) {
            int n = nt * 16 + col;
            if (ok) atomicAdd(&agg[(long)d * HD + n], acc2[nt][i] + b2[n]);
        }
    }
}

// ---------------- node update: h += relu(concat(h, extra/cnt) @ W + b) ----------------
__global__ __launch_bounds__(256, 1) void node_update_kernel(
    float* __restrict__ h, const float* __restrict__ extra, const float* __restrict__ cnt,
    const v8u* __restrict__ Wf, const float* __restrict__ bias, int N) {
    __shared__ unsigned int sA[WAVES][256 * 8];
    const int wave = threadIdx.x >> 5, lane = threadIdx.x & 31;
    const int tile = blockIdx.x * WAVES + wave;
    const long n0  = (long)tile * 16;
    const bool active = n0 < N;
    __builtin_prefetch(Wf, 0, 1);
    if (active) {
        for (int t = lane; t < 16 * 64; t += 32) {
            int row = t >> 6, c4 = (t & 63) << 2;
            long r = n0 + row; if (r >= N) r = N - 1;
            float4 v;
            if (c4 < HD) {
                v = *(const float4*)&h[r * HD + c4];
            } else {
                float cc = cnt[r]; cc = cc > 1.f ? cc : 1.f;
                v = *(const float4*)&extra[r * HD + (c4 - HD)];
                v.x /= cc; v.y /= cc; v.z /= cc; v.w /= cc;
            }
            unsigned int p0 = (unsigned int)f32_to_bf16(v.x) | ((unsigned int)f32_to_bf16(v.y) << 16);
            unsigned int p1 = (unsigned int)f32_to_bf16(v.z) | ((unsigned int)f32_to_bf16(v.w) << 16);
            int fi = frag_idx(row, c4);
            *(uint2*)&sA[wave][fi] = make_uint2(p0, p1);
        }
    }
    __syncthreads();
    if (!active) return;
    v8f acc[8] = {};
    wmma_frag_tile<256>(sA[wave], Wf, lane, acc);
    const int col = lane & 15, half = lane >> 4;
#pragma unroll
    for (int i = 0; i < 8; ++i) {
        long m = n0 + i + 8 * half;
        bool ok = m < N;
#pragma unroll
        for (int nt = 0; nt < 8; ++nt) {
            int n = nt * 16 + col;
            if (ok) {
                float v = acc[nt][i] + bias[n];
                v = v > 0.f ? v : 0.f;
                h[m * HD + n] += v;
            }
        }
    }
}

// ---------------- spacepoint MLP: sp = relu(spacc @ W_sp + b_sp), K=128 ----------------
__global__ __launch_bounds__(256, 1) void sp_mlp_kernel(
    const float* __restrict__ spacc, const v8u* __restrict__ Wf,
    const float* __restrict__ bias, float* __restrict__ sp, int N) {
    __shared__ unsigned int sA[WAVES][128 * 8];
    const int wave = threadIdx.x >> 5, lane = threadIdx.x & 31;
    const int tile = blockIdx.x * WAVES + wave;
    const long n0  = (long)tile * 16;
    const bool active = n0 < N;
    if (active) {
        for (int t = lane; t < 16 * 32; t += 32) {
            int row = t >> 5, c4 = (t & 31) << 2;
            long r = n0 + row; if (r >= N) r = N - 1;
            float4 v = *(const float4*)&spacc[r * HD + c4];
            unsigned int p0 = (unsigned int)f32_to_bf16(v.x) | ((unsigned int)f32_to_bf16(v.y) << 16);
            unsigned int p1 = (unsigned int)f32_to_bf16(v.z) | ((unsigned int)f32_to_bf16(v.w) << 16);
            int fi = frag_idx(row, c4);
            *(uint2*)&sA[wave][fi] = make_uint2(p0, p1);
        }
    }
    __syncthreads();
    if (!active) return;
    v8f acc[8] = {};
    wmma_frag_tile<128>(sA[wave], Wf, lane, acc);
    const int col = lane & 15, half = lane >> 4;
#pragma unroll
    for (int i = 0; i < 8; ++i) {
        long m = n0 + i + 8 * half;
        bool ok = m < N;
#pragma unroll
        for (int nt = 0; nt < 8; ++nt) {
            int n = nt * 16 + col;
            if (ok) {
                float v = acc[nt][i] + bias[n];
                sp[m * HD + n] = v > 0.f ? v : 0.f;
            }
        }
    }
}

// ---------------- row scatter-add (float4 per thread): dst[dstIdx[e]] += src[srcIdx[e]] ----
__global__ void scatter_rows_kernel(const float* __restrict__ src, const int* __restrict__ srcIdx,
                                    const int* __restrict__ dstIdx, float* __restrict__ dst, int M) {
    long i = (long)blockIdx.x * blockDim.x + threadIdx.x;
    long total = (long)M * (HD / 4);
    if (i >= total) return;
    int e = (int)(i / (HD / 4)), c4 = (int)(i % (HD / 4)) * 4;
    const float4 v = *(const float4*)&src[(long)srcIdx[e] * HD + c4];
    float* dp = &dst[(long)dstIdx[e] * HD + c4];
    atomicAdd(dp + 0, v.x); atomicAdd(dp + 1, v.y);
    atomicAdd(dp + 2, v.z); atomicAdd(dp + 3, v.w);
}

// ---------------- acc += sum / max(cnt,1) per row (float4 per thread) ----------------
__global__ void mean_add_kernel(float* __restrict__ acc, const float* __restrict__ sum,
                                const float* __restrict__ cnt, int N) {
    long i = (long)blockIdx.x * blockDim.x + threadIdx.x;
    if (i >= (long)N * (HD / 4)) return;
    long r = i / (HD / 4);
    long idx = r * HD + (i % (HD / 4)) * 4;
    float c = cnt[r]; c = c > 1.f ? c : 1.f;
    float4 s = *(const float4*)&sum[idx];
    float4 a = *(float4*)&acc[idx];
    a.x += s.x / c; a.y += s.y / c; a.z += s.z / c; a.w += s.w / c;
    *(float4*)&acc[idx] = a;
}

// ---------------- decoder: out = h @ W_sem + b_sem ----------------
__global__ void decoder_kernel(const float* __restrict__ h, const float* __restrict__ W,
                               const float* __restrict__ b, float* __restrict__ out, int N) {
    int i = blockIdx.x * blockDim.x + threadIdx.x;
    if (i >= N * NCLASS) return;
    int n = i / NCLASS, c = i % NCLASS;
    float a = b[c];
    for (int k = 0; k < HD; ++k) a += h[(long)n * HD + k] * W[k * NCLASS + c];
    out[i] = a;
}

// =====================================================================

extern "C" void kernel_launch(void* const* d_in, const int* in_sizes, int n_in,
                              void* d_out, int out_size, void* d_ws, size_t ws_size,
                              hipStream_t stream) {
    (void)in_sizes; (void)n_in; (void)out_size; (void)ws_size;
    const float* x[3]     = {(const float*)d_in[0], (const float*)d_in[3], (const float*)d_in[6]};
    const int*   edge[3]  = {(const int*)d_in[1],   (const int*)d_in[4],   (const int*)d_in[7]};
    const int*   nexus[3] = {(const int*)d_in[2],   (const int*)d_in[5],   (const int*)d_in[8]};
    const float* W_enc  = (const float*)d_in[9];   const float* b_enc  = (const float*)d_in[10];
    const float* W_msg1 = (const float*)d_in[11];  const float* b_msg1 = (const float*)d_in[12];
    const float* W_msg2 = (const float*)d_in[13];  const float* b_msg2 = (const float*)d_in[14];
    const float* W_upd  = (const float*)d_in[15];  const float* b_upd  = (const float*)d_in[16];
    const float* W_sp   = (const float*)d_in[17];  const float* b_sp   = (const float*)d_in[18];
    const float* W_nx   = (const float*)d_in[19];  const float* b_nx   = (const float*)d_in[20];
    const float* W_sem  = (const float*)d_in[21];  const float* b_sem  = (const float*)d_in[22];
    float* out = (float*)d_out;

    // ---- carve workspace ----
    size_t off = 0;
    auto carve = [&](size_t bytes) {
        off = (off + 255) & ~(size_t)255;
        void* p = (char*)d_ws + off;
        off += bytes;
        return p;
    };
    const size_t hitRowBytes = (size_t)NHIT * HD * sizeof(float);
    const size_t spRowBytes  = (size_t)NSP  * HD * sizeof(float);
    float* h[3];  for (int p = 0; p < 3; ++p) h[p] = (float*)carve(hitRowBytes);
    float* aux   = (float*)carve(hitRowBytes);
    float* spsum = (float*)carve(spRowBytes);
    float* spacc = (float*)carve(spRowBytes);
    float* sp    = (float*)carve(spRowBytes);
    float* cntE[3], *fcnt[3], *bcnt[3];
    for (int p = 0; p < 3; ++p) cntE[p] = (float*)carve((size_t)NHIT * sizeof(float));
    for (int p = 0; p < 3; ++p) fcnt[p] = (float*)carve((size_t)NSP  * sizeof(float));
    for (int p = 0; p < 3; ++p) bcnt[p] = (float*)carve((size_t)NHIT * sizeof(float));
    v8u* W1f  = (v8u*)carve((size_t)256 * 64 * 4);
    v8u* W2f  = (v8u*)carve((size_t)128 * 64 * 4);
    v8u* Wupf = (v8u*)carve((size_t)256 * 64 * 4);
    v8u* Wspf = (v8u*)carve((size_t)128 * 64 * 4);
    v8u* Wnxf = (v8u*)carve((size_t)256 * 64 * 4);

    const int B = 256;
    auto g = [](long n, int b) { return (int)((n + b - 1) / b); };

    // ---- pack weights to bf16 fragment-major layout ----
    pack_w_kernel<<<g(256 * 64, B), B, 0, stream>>>(W_msg1, (unsigned int*)W1f, 256);
    pack_w_kernel<<<g(128 * 64, B), B, 0, stream>>>(W_msg2, (unsigned int*)W2f, 128);
    pack_w_kernel<<<g(256 * 64, B), B, 0, stream>>>(W_upd,  (unsigned int*)Wupf, 256);
    pack_w_kernel<<<g(128 * 64, B), B, 0, stream>>>(W_sp,   (unsigned int*)Wspf, 128);
    pack_w_kernel<<<g(256 * 64, B), B, 0, stream>>>(W_nx,   (unsigned int*)Wnxf, 256);

    // ---- segment counts (graph structure is constant over iterations) ----
    for (int p = 0; p < 3; ++p) {
        hipMemsetAsync(cntE[p], 0, (size_t)NHIT * 4, stream);
        hipMemsetAsync(fcnt[p], 0, (size_t)NSP  * 4, stream);
        hipMemsetAsync(bcnt[p], 0, (size_t)NHIT * 4, stream);
        count_kernel<<<g(NEDGE, B), B, 0, stream>>>(edge[p] + NEDGE, cntE[p], NEDGE);  // dst
        count_kernel<<<g(NSP,   B), B, 0, stream>>>(nexus[p] + NSP,  fcnt[p], NSP);    // nx[1]
        count_kernel<<<g(NSP,   B), B, 0, stream>>>(nexus[p],        bcnt[p], NSP);    // nx[0]
    }

    // ---- encoder ----
    for (int p = 0; p < 3; ++p)
        encoder_kernel<<<g((long)NHIT * HD, B), B, 0, stream>>>(x[p], W_enc, b_enc, h[p], NHIT);

    const int edgeTiles = (NEDGE + 15) / 16;
    const int hitTiles  = (NHIT  + 15) / 16;
    const int spTiles   = (NSP   + 15) / 16;
    const int edgeGrid  = (edgeTiles + WAVES - 1) / WAVES;
    const int hitGrid   = (hitTiles  + WAVES - 1) / WAVES;
    const int spGrid    = (spTiles   + WAVES - 1) / WAVES;

    // ---- core message-passing iterations ----
    for (int it = 0; it < NITERS; ++it) {
        // planar convolution per plane
        for (int p = 0; p < 3; ++p) {
            hipMemsetAsync(aux, 0, hitRowBytes, stream);
            edge_mlp_kernel<<<edgeGrid, B, 0, stream>>>(h[p], edge[p], W1f, b_msg1, W2f, b_msg2,
                                                        aux, NEDGE);
            node_update_kernel<<<hitGrid, B, 0, stream>>>(h[p], aux, cntE[p], Wupf, b_upd, NHIT);
        }
        // nexus: planes -> spacepoints
        hipMemsetAsync(spacc, 0, spRowBytes, stream);
        for (int p = 0; p < 3; ++p) {
            hipMemsetAsync(spsum, 0, spRowBytes, stream);
            scatter_rows_kernel<<<g((long)NSP * (HD / 4), B), B, 0, stream>>>(
                h[p], nexus[p], nexus[p] + NSP, spsum, NSP);
            mean_add_kernel<<<g((long)NSP * (HD / 4), B), B, 0, stream>>>(spacc, spsum, fcnt[p], NSP);
        }
        sp_mlp_kernel<<<spGrid, B, 0, stream>>>(spacc, Wspf, b_sp, sp, NSP);
        // nexus: spacepoints -> planes
        for (int p = 0; p < 3; ++p) {
            hipMemsetAsync(aux, 0, hitRowBytes, stream);
            scatter_rows_kernel<<<g((long)NSP * (HD / 4), B), B, 0, stream>>>(
                sp, nexus[p] + NSP, nexus[p], aux, NSP);
            node_update_kernel<<<hitGrid, B, 0, stream>>>(h[p], aux, bcnt[p], Wnxf, b_nx, NHIT);
        }
    }

    // ---- decoder ----
    for (int p = 0; p < 3; ++p)
        decoder_kernel<<<g((long)NHIT * NCLASS, B), B, 0, stream>>>(h[p], W_sem, b_sem,
                                                                    out + (size_t)p * NHIT * NCLASS,
                                                                    NHIT);
}
